// Attention_70317204570333
// MI455X (gfx1250) — compile-verified, matrix operand order checked by reading, not run on hardware
//
#include <hip/hip_runtime.h>
#include <hip/hip_bf16.h>

// ---------------------------------------------------------------------------
// Types
// ---------------------------------------------------------------------------
typedef __bf16 bf16_t;
typedef __attribute__((ext_vector_type(16))) __bf16 v16bf;
typedef __attribute__((ext_vector_type(8)))  __bf16 v8bf;
typedef __attribute__((ext_vector_type(8)))  float   v8f;

#define EMBED   768
#define NHEAD   12
#define HD      64
#define SEQ     4096
#define QKV_LD  2304   // 3 * EMBED

// fp32 -> bf16: native conversion (gfx1250 has bf16 cvt in hardware)
__device__ __forceinline__ bf16_t f2bf(float f) { return static_cast<bf16_t>(f); }

// LDS byte address for async-to-LDS ops: generic AS pointer to __shared__
// carries the LDS offset in addr[31:0] (ISA 10.2 aperture mapping).
__device__ __forceinline__ unsigned lds_addr(const void* p) {
    return (unsigned)(uintptr_t)p;
}

// GLOBAL_LOAD_ASYNC_TO_LDS_B128, GVS mode: mem = SGPR64(base) + VGPR32(offset).
// Tracked by ASYNCcnt; data flows DMA-style into LDS without touching VGPRs.
__device__ __forceinline__ void async_b128(unsigned lds_byte, const void* base,
                                           unsigned goff_byte) {
    asm volatile("global_load_async_to_lds_b128 %0, %1, %2"
                 :: "v"(lds_byte), "v"(goff_byte), "s"(base)
                 : "memory");
}

__device__ __forceinline__ void wait_async0() {
#if __has_builtin(__builtin_amdgcn_s_wait_asynccnt)
    __builtin_amdgcn_s_wait_asynccnt(0);
#else
    asm volatile("s_wait_asynccnt 0x0" ::: "memory");
#endif
}

union Frag16 { v16bf v; v8bf h[2]; };

// Load a 16x32 bf16 operand fragment (A-layout; B via transposed-in-LDS).
// Per ISA 7.12.2: lane half h = lane>>4, row = lane&15,
//   elements 0..7  = K[h*8 .. h*8+7], elements 8..15 = K[16+h*8 .. +7].
__device__ __forceinline__ v16bf load_frag16(const bf16_t* base, int ld,
                                             int row0, int k0, int lane) {
    int hh = lane >> 4;
    int r  = lane & 15;
    const bf16_t* p = base + (size_t)(row0 + r) * ld + k0 + hh * 8;
    Frag16 f;
    f.h[0] = *(const v8bf*)(p);
    f.h[1] = *(const v8bf*)(p + 16);
    return f.v;
}

__device__ __forceinline__ v8f wmma_bf16(v16bf a, v16bf b, v8f c) {
    return __builtin_amdgcn_wmma_f32_16x16x32_bf16(
        false, a, false, b, (short)0, c, false, false);
}

// ---------------------------------------------------------------------------
// Generic tiled GEMM:  C[M,N] = A[M,K] * B[K,N] + bias[N]
//   A: fp32 or bf16 (row-major, lda); B: fp32 (row-major, ldb)
//   C: fp32 or bf16 (row-major, ldc)
// Block tile 128x128, 256 threads = 8 waves (2x4 grid of 64x32 wave tiles).
// Depth-1 double-buffered pipeline: next tile's loads (async DMA for bf16 A)
// overlap the current tile's WMMAs; one barrier per k-tile.
// ---------------------------------------------------------------------------
template<bool A_BF16, bool OUT_F32>
__global__ void __launch_bounds__(256)
gemm_wmma(const void* __restrict__ Ap, const float* __restrict__ B,
          const float* __restrict__ bias, void* __restrict__ Cp,
          int K, int lda, int ldb, int ldc) {
    __shared__ bf16_t As[2][128 * 32];   // [m][k]
    __shared__ bf16_t Bs[2][128 * 32];   // transposed: [n][k]

    const int tid  = threadIdx.x;
    const int lane = tid & 31;
    const int wave = tid >> 5;
    const int wrow = wave >> 2;       // 0..1 -> 64 rows each
    const int wcol = wave & 3;        // 0..3 -> 32 cols each
    const int hh   = lane >> 4;
    const int l15  = lane & 15;

    const int bm = blockIdx.y * 128;
    const int bn = blockIdx.x * 128;

    v8f acc[4][2];
#pragma unroll
    for (int mt = 0; mt < 4; ++mt)
#pragma unroll
        for (int nt = 0; nt < 2; ++nt) acc[mt][nt] = 0.0f;

    const int arow  = tid >> 1;     // A staging: 2 threads per 128-row tile row
    const int ahalf = tid & 1;
    const int br    = tid >> 3;     // B staging: k-row 0..31
    const int bseg  = tid & 7;      // 16-col segment

    // ---- prologue: stage tile 0 into buffer 0 ----
    {
        if (A_BF16) {
            unsigned l = lds_addr(&As[0][arow * 32 + ahalf * 16]);
            unsigned g = (unsigned)(((size_t)(bm + arow) * lda + ahalf * 16) * 2);
            async_b128(l, Ap, g);
            async_b128(l + 16, Ap, g + 16);
        } else {
            const float* a = (const float*)Ap + (size_t)(bm + arow) * lda + ahalf * 16;
            float t[16];
            *(float4*)&t[0]  = *(const float4*)(a);
            *(float4*)&t[4]  = *(const float4*)(a + 4);
            *(float4*)&t[8]  = *(const float4*)(a + 8);
            *(float4*)&t[12] = *(const float4*)(a + 12);
            bf16_t* dst = &As[0][arow * 32 + ahalf * 16];
            v8bf y0, y1;
#pragma unroll
            for (int j = 0; j < 8; ++j) { y0[j] = f2bf(t[j]); y1[j] = f2bf(t[8 + j]); }
            *(v8bf*)(dst) = y0;  *(v8bf*)(dst + 8) = y1;
        }
        const float* bp = B + (size_t)br * ldb + bn + bseg * 16;
        float t[16];
        *(float4*)&t[0]  = *(const float4*)(bp);
        *(float4*)&t[4]  = *(const float4*)(bp + 4);
        *(float4*)&t[8]  = *(const float4*)(bp + 8);
        *(float4*)&t[12] = *(const float4*)(bp + 12);
#pragma unroll
        for (int j = 0; j < 16; ++j)
            Bs[0][(size_t)(bseg * 16 + j) * 32 + br] = f2bf(t[j]);
        if (A_BF16) wait_async0();
    }
    __syncthreads();

    int cur = 0;
    for (int k0 = 0; k0 < K; k0 += 32) {
        const int nxt = cur ^ 1;
        const bool has_next = (k0 + 32) < K;

        // ---- issue next tile's global traffic (overlaps WMMAs below) ----
        float ta[16], tb[16];
        if (has_next) {
            const int kn = k0 + 32;
            if (A_BF16) {
                unsigned l = lds_addr(&As[nxt][arow * 32 + ahalf * 16]);
                unsigned g = (unsigned)(((size_t)(bm + arow) * lda + kn + ahalf * 16) * 2);
                async_b128(l, Ap, g);
                async_b128(l + 16, Ap, g + 16);
            } else {
                const float* a = (const float*)Ap + (size_t)(bm + arow) * lda + kn + ahalf * 16;
                *(float4*)&ta[0]  = *(const float4*)(a);
                *(float4*)&ta[4]  = *(const float4*)(a + 4);
                *(float4*)&ta[8]  = *(const float4*)(a + 8);
                *(float4*)&ta[12] = *(const float4*)(a + 12);
            }
            const float* bp = B + (size_t)(kn + br) * ldb + bn + bseg * 16;
            *(float4*)&tb[0]  = *(const float4*)(bp);
            *(float4*)&tb[4]  = *(const float4*)(bp + 4);
            *(float4*)&tb[8]  = *(const float4*)(bp + 8);
            *(float4*)&tb[12] = *(const float4*)(bp + 12);
            if (k0 + 64 < K)   // warm L2 for the tile after next
                __builtin_prefetch(bp + 32 * ldb, 0, 3);
        }

        // ---- compute from current buffers ----
        v16bf bfr0 = load_frag16(Bs[cur], 32, wcol * 32,      0, lane);
        v16bf bfr1 = load_frag16(Bs[cur], 32, wcol * 32 + 16, 0, lane);
#pragma unroll
        for (int mt = 0; mt < 4; ++mt) {
            v16bf af = load_frag16(As[cur], 32, wrow * 64 + mt * 16, 0, lane);
            acc[mt][0] = wmma_bf16(af, bfr0, acc[mt][0]);
            acc[mt][1] = wmma_bf16(af, bfr1, acc[mt][1]);
        }

        // ---- convert + store next tile into the other buffer ----
        if (has_next) {
            if (!A_BF16) {
                bf16_t* dst = &As[nxt][arow * 32 + ahalf * 16];
                v8bf y0, y1;
#pragma unroll
                for (int j = 0; j < 8; ++j) { y0[j] = f2bf(ta[j]); y1[j] = f2bf(ta[8 + j]); }
                *(v8bf*)(dst) = y0;  *(v8bf*)(dst + 8) = y1;
            }
#pragma unroll
            for (int j = 0; j < 16; ++j)
                Bs[nxt][(size_t)(bseg * 16 + j) * 32 + br] = f2bf(tb[j]);
            if (A_BF16) wait_async0();
        }
        __syncthreads();
        cur = nxt;
    }

    // ---- epilogue: C/D layout: elem i -> row = i + 8*hh, col = l15 ----
#pragma unroll
    for (int mt = 0; mt < 4; ++mt)
#pragma unroll
        for (int nt = 0; nt < 2; ++nt) {
            int col = bn + wcol * 32 + nt * 16 + l15;
            float bv = bias[col];
#pragma unroll
            for (int i = 0; i < 8; ++i) {
                int row = bm + wrow * 64 + mt * 16 + hh * 8 + i;
                float v = acc[mt][nt][i] + bv;
                if (OUT_F32)
                    ((float*)Cp)[(size_t)row * ldc + col] = v;
                else
                    ((bf16_t*)Cp)[(size_t)row * ldc + col] = f2bf(v);
            }
        }
}

// ---------------------------------------------------------------------------
// Flash attention: per block = one (b, h, 128-row Q block).
// Streams 64-key K/V blocks with depth-1 double buffering: next block's
// K async-DMA and V global loads overlap the current block's WMMAs/softmax.
// ---------------------------------------------------------------------------
__global__ void __launch_bounds__(256)
flash_attn(const void* __restrict__ qkvp, void* __restrict__ outp) {
    const bf16_t* qkv = (const bf16_t*)qkvp;
    bf16_t* attn      = (bf16_t*)outp;

    __shared__ bf16_t Qs[128 * 64];        // [qrow][d]
    __shared__ bf16_t Ks[2][64 * 64];      // [key][d]
    __shared__ bf16_t Vs[2][64 * 64];      // transposed: [d][key]
    __shared__ bf16_t Ps[8 * 16 * 64];     // per-wave P strip [row][key]

    const int tid  = threadIdx.x;
    const int lane = tid & 31;
    const int wave = tid >> 5;
    const int hh   = lane >> 4;
    const int l15  = lane & 15;

    const int bid = blockIdx.x;
    const int qb  = bid & 31;              // 4096/128 = 32 q blocks
    const int h   = (bid >> 5) % NHEAD;
    const int b   = bid / (32 * NHEAD);

    const size_t rowbase = (size_t)b * SEQ;
    const bf16_t* Qg = qkv + rowbase * QKV_LD + h * HD;
    const bf16_t* Kg = Qg + EMBED;
    const bf16_t* Vg = Qg + 2 * EMBED;

    const int krow = tid >> 2, kseg = tid & 3;

    // ---- stage Q block via async DMA + kv block 0 ----
    {
        int row = tid >> 1, half = tid & 1;
        unsigned l = lds_addr(&Qs[row * 64 + half * 32]);
        unsigned g = (unsigned)(((size_t)(qb * 128 + row) * QKV_LD + half * 32) * 2);
#pragma unroll
        for (int j = 0; j < 4; ++j)
            async_b128(l + j * 16, Qg, g + j * 16);

        unsigned lk = lds_addr(&Ks[0][krow * 64 + kseg * 16]);
        unsigned gk = (unsigned)(((size_t)krow * QKV_LD + kseg * 16) * 2);
        async_b128(lk,      Kg, gk);
        async_b128(lk + 16, Kg, gk + 16);

        const bf16_t* src = Vg + (size_t)krow * QKV_LD + kseg * 16;
        v8bf a0 = *(const v8bf*)(src);
        v8bf a1 = *(const v8bf*)(src + 8);
#pragma unroll
        for (int j = 0; j < 8; ++j) {
            Vs[0][(size_t)(kseg * 16 + j)     * 64 + krow] = a0[j];
            Vs[0][(size_t)(kseg * 16 + 8 + j) * 64 + krow] = a1[j];
        }
    }
    wait_async0();
    __syncthreads();

    v16bf qa0 = load_frag16(Qs, 64, wave * 16, 0,  lane);
    v16bf qa1 = load_frag16(Qs, 64, wave * 16, 32, lane);

    float mrow[8], lrow[8];
    v8f o[4];
#pragma unroll
    for (int i = 0; i < 8; ++i) { mrow[i] = -3.0e38f; lrow[i] = 0.0f; }
#pragma unroll
    for (int nt = 0; nt < 4; ++nt) o[nt] = 0.0f;

    const float SCALE = 0.125f;   // 64^-0.5
    int cur = 0;

    for (int kb = 0; kb < SEQ / 64; ++kb) {
        const int nxt = cur ^ 1;
        const bool has_next = (kb + 1) < SEQ / 64;

        // ---- issue next kv block's traffic (overlaps compute below) ----
        v8bf va0 = (bf16_t)0.0f, va1 = (bf16_t)0.0f;
        if (has_next) {
            const int key0n = (kb + 1) * 64;
            unsigned lk = lds_addr(&Ks[nxt][krow * 64 + kseg * 16]);
            unsigned gk = (unsigned)(((size_t)(key0n + krow) * QKV_LD + kseg * 16) * 2);
            async_b128(lk,      Kg, gk);
            async_b128(lk + 16, Kg, gk + 16);
            const bf16_t* src = Vg + (size_t)(key0n + krow) * QKV_LD + kseg * 16;
            va0 = *(const v8bf*)(src);
            va1 = *(const v8bf*)(src + 8);
        }

        // ---- S strip = Q_strip (16x64) @ K^T (64 keys) ----
        const bf16_t* Kc = Ks[cur];
        const bf16_t* Vc = Vs[cur];
        v8f s[4];
#pragma unroll
        for (int nt = 0; nt < 4; ++nt) {
            v8f z = 0.0f;
            z = wmma_bf16(qa0, load_frag16(Kc, 64, nt * 16, 0,  lane), z);
            z = wmma_bf16(qa1, load_frag16(Kc, 64, nt * 16, 32, lane), z);
            s[nt] = z;
        }

        // ---- online softmax (row r = hh*8 + i lives across 16 lanes) ----
        float alpha[8];
#pragma unroll
        for (int i = 0; i < 8; ++i) {
            float mx = -3.0e38f;
#pragma unroll
            for (int nt = 0; nt < 4; ++nt) {
                s[nt][i] *= SCALE;
                mx = fmaxf(mx, s[nt][i]);
            }
#pragma unroll
            for (int d = 1; d < 16; d <<= 1)
                mx = fmaxf(mx, __shfl_xor(mx, d, 32));
            float mn = fmaxf(mrow[i], mx);
            alpha[i] = __expf(mrow[i] - mn);
            float ls = 0.0f;
#pragma unroll
            for (int nt = 0; nt < 4; ++nt) {
                float p = __expf(s[nt][i] - mn);
                s[nt][i] = p;
                ls += p;
            }
#pragma unroll
            for (int d = 1; d < 16; d <<= 1)
                ls += __shfl_xor(ls, d, 32);
            lrow[i] = lrow[i] * alpha[i] + ls;
            mrow[i] = mn;
        }

        // ---- C-layout -> A-layout transpose of P via per-wave LDS ----
        bf16_t* Pw = &Ps[wave * 16 * 64];
#pragma unroll
        for (int nt = 0; nt < 4; ++nt)
#pragma unroll
            for (int i = 0; i < 8; ++i)
                Pw[(size_t)(hh * 8 + i) * 64 + nt * 16 + l15] = f2bf(s[nt][i]);

        // ---- rescale running O ----
#pragma unroll
        for (int nt = 0; nt < 4; ++nt)
#pragma unroll
            for (int i = 0; i < 8; ++i) o[nt][i] *= alpha[i];

        // ---- O += P (16x64) @ V (64 keys x 64 d) ----
        v16bf pa0 = load_frag16(Pw, 64, 0, 0,  lane);
        v16bf pa1 = load_frag16(Pw, 64, 0, 32, lane);
#pragma unroll
        for (int nt = 0; nt < 4; ++nt) {
            o[nt] = wmma_bf16(pa0, load_frag16(Vc, 64, nt * 16, 0,  lane), o[nt]);
            o[nt] = wmma_bf16(pa1, load_frag16(Vc, 64, nt * 16, 32, lane), o[nt]);
        }

        // ---- commit next V tile (transposed scatter), close the pipe ----
        if (has_next) {
#pragma unroll
            for (int j = 0; j < 8; ++j) {
                Vs[nxt][(size_t)(kseg * 16 + j)     * 64 + krow] = va0[j];
                Vs[nxt][(size_t)(kseg * 16 + 8 + j) * 64 + krow] = va1[j];
            }
            wait_async0();
        }
        __syncthreads();
        cur = nxt;
    }

    // ---- epilogue: normalize and write bf16 [B*N, EMBED] ----
#pragma unroll
    for (int i = 0; i < 8; ++i) {
        float inv = 1.0f / lrow[i];
        size_t row = rowbase + (size_t)qb * 128 + wave * 16 + hh * 8 + i;
#pragma unroll
        for (int nt = 0; nt < 4; ++nt) {
            int col = h * HD + nt * 16 + l15;
            attn[row * EMBED + col] = f2bf(o[nt][i] * inv);
        }
    }
}

// ---------------------------------------------------------------------------
// Host launcher
// ---------------------------------------------------------------------------
extern "C" void kernel_launch(void* const* d_in, const int* in_sizes, int n_in,
                              void* d_out, int out_size, void* d_ws, size_t ws_size,
                              hipStream_t stream) {
    (void)in_sizes; (void)n_in; (void)out_size; (void)ws_size;
    const float* x     = (const float*)d_in[0];   // [2, 4096, 768]
    const float* Wqkv  = (const float*)d_in[1];   // [768, 2304]
    const float* bqkv  = (const float*)d_in[2];   // [2304]
    const float* Wproj = (const float*)d_in[3];   // [768, 768]
    const float* bproj = (const float*)d_in[4];   // [768]
    float* out = (float*)d_out;                   // [2, 4096, 768] fp32

    char* ws = (char*)d_ws;
    void* qkv_bf  = ws;                                        // 8192*2304 bf16
    void* attn_bf = ws + (size_t)8192 * QKV_LD * 2;            // 8192*768 bf16

    // 1) QKV projection: [8192,768] @ [768,2304] + b -> bf16 qkv
    gemm_wmma<false, false><<<dim3(QKV_LD / 128, 8192 / 128), 256, 0, stream>>>(
        x, Wqkv, bqkv, qkv_bf, EMBED, EMBED, QKV_LD, QKV_LD);

    // 2) Flash attention: 2 batches * 12 heads * 32 q-blocks = 768 blocks
    flash_attn<<<dim3(2 * NHEAD * (SEQ / 128)), 256, 0, stream>>>(qkv_bf, attn_bf);

    // 3) Output projection: [8192,768] @ [768,768] + b -> fp32 out
    gemm_wmma<true, true><<<dim3(EMBED / 128, 8192 / 128), 256, 0, stream>>>(
        attn_bf, Wproj, bproj, out, EMBED, EMBED, EMBED, EMBED);
}
